// RoIAlignAppearanceEncoder_89266600280310
// MI455X (gfx1250) — matrix operand order repaired and checked by reading, not compiled, not run on hardware
//
#include <hip/hip_runtime.h>
#include <hip/hip_bf16.h>

typedef __attribute__((ext_vector_type(16))) _Float16 v16h;
typedef __attribute__((ext_vector_type(8)))  _Float16 v8h;
typedef __attribute__((ext_vector_type(8)))  float    v8f;

// Problem dims (fixed by the reference)
#define BB    16
#define HH    512
#define WW    512
#define NBOX  16
#define CC    256
#define DD    256
#define FH    32
#define FW    32
#define KCONV 768          // 16*16*3
#define MCONV 16384        // B*32*32
#define SROI  7

// ---------------------------------------------------------------------------
// im2col: images [B,512,512,3] f32 -> A [MCONV, KCONV] f16
// row m = (b, oy, ox); col k = (ky*16+kx)*3 + ci  (matches HWIO weight order)
// ---------------------------------------------------------------------------
__global__ __launch_bounds__(256) void k_im2col(const float* __restrict__ img,
                                                _Float16* __restrict__ A) {
  long idx = (long)blockIdx.x * blockDim.x + threadIdx.x;
  if (idx >= (long)MCONV * KCONV) return;
  int k = (int)(idx % KCONV);
  int m = (int)(idx / KCONV);
  int ci = k % 3;
  int kx = (k / 3) & 15;
  int ky = k / 48;
  int ox = m & 31, oy = (m >> 5) & 31, b = m >> 10;
  int y = oy * 16 + ky, x = ox * 16 + kx;
  A[idx] = (_Float16)img[(((long)b * HH + y) * WW + x) * 3 + ci];
}

// conv_w [768,256] f32 (K-major) -> BT [256,768] f16 (N-major, K contiguous)
__global__ __launch_bounds__(256) void k_conv_w_t(const float* __restrict__ w,
                                                  _Float16* __restrict__ BT) {
  int idx = blockIdx.x * blockDim.x + threadIdx.x;  // co*KCONV + k
  if (idx >= CC * KCONV) return;
  int k = idx % KCONV, co = idx / KCONV;
  BT[idx] = (_Float16)w[(long)k * CC + co];
}

// dense_w [256,256] f32 (C-major) -> DT [256,256] f16 (D-major, C contiguous)
__global__ __launch_bounds__(256) void k_dense_w_t(const float* __restrict__ w,
                                                   _Float16* __restrict__ DT) {
  int idx = blockIdx.x * blockDim.x + threadIdx.x;  // d*CC + c
  if (idx >= DD * CC) return;
  int c = idx % CC, d = idx / CC;
  DT[idx] = (_Float16)w[(long)c * DD + d];
}

// Two contiguous 16B loads -> one 16-half WMMA fragment register block.
__device__ __forceinline__ v16h load_frag16(const _Float16* p0,
                                            const _Float16* p1) {
  v8h lo = *(const v8h*)p0;
  v8h hi = *(const v8h*)p1;
  v16h f;
#pragma unroll
  for (int i = 0; i < 8; ++i) { f[i] = lo[i]; f[8 + i] = hi[i]; }
  return f;
}

template <int MT, int NT>
struct FragSet {
  v16h a[MT];
  v16h b[NT];
};

// ---------------------------------------------------------------------------
// WMMA GEMM: C[M, Ncols] = A[M, KDIM](f16) * BT[Ncols, KDIM]^T(f16) + bias
// Register-blocked: one wave owns MT 16-row M-tiles x NT 16-col N-tiles
// (MT*NT WMMAs per K-step vs MT+NT fragment loads). The K-loop is
// software-pipelined with an explicit PING-PONG over two fragment sets
// (no register copies): each iteration issues loads into one set while the
// WMMAs consume the other, so loads have a full K-step of matrix work to
// complete and the only ordering constraint is the WAR on the refilled set.
//
// Fragment layouts (wave32, per CDNA5 ISA 7.12.2):
//  A 16x32 f16 : lane L -> row r = L%16, half h = L/16
//     a[0..7]  = A[r][k0 + h*8 + 0..7]        (one 16B load)
//     a[8..15] = A[r][k0 + 16 + h*8 + 0..7]   (one 16B load)
//  B 32x16 f16 : lane L -> col n = L%16, half h = L/16
//     b[0..7]  = BT[n][k0 + h*8 + 0..7], b[8..15] = BT[n][k0 + 16 + h*8 + 0..7]
//  C 16x16 f32 : lane L -> col n = L%16, rows (L/16)*8 + i for acc[i]
// ---------------------------------------------------------------------------
template <int KDIM, int MT, int NT, bool RELU>
__global__ __launch_bounds__(256) void k_gemm(const _Float16* __restrict__ A,
                                              const _Float16* __restrict__ BT,
                                              const float* __restrict__ bias,
                                              float* __restrict__ C,
                                              int MGroups, int NGroups,
                                              int Ncols) {
  const int lane  = threadIdx.x & 31;
  const int wflat = (int)((blockIdx.x * blockDim.x + threadIdx.x) >> 5);
  const int mg = wflat / NGroups;
  const int g  = wflat % NGroups;
  if (mg >= MGroups) return;

  const int r = lane & 15;
  const int h = lane >> 4;

  const _Float16* Arow[MT];
#pragma unroll
  for (int mi = 0; mi < MT; ++mi)
    Arow[mi] = A + ((size_t)((mg * MT + mi) * 16 + r)) * KDIM;
  const _Float16* Brow[NT];
#pragma unroll
  for (int t = 0; t < NT; ++t)
    Brow[t] = BT + ((size_t)((g * NT + t) * 16 + r)) * KDIM;

  v8f acc[MT][NT] = {};

  auto LOAD = [&](FragSet<MT, NT>& F, int k0) {
#pragma unroll
    for (int mi = 0; mi < MT; ++mi)
      F.a[mi] =
          load_frag16(Arow[mi] + k0 + h * 8, Arow[mi] + k0 + 16 + h * 8);
#pragma unroll
    for (int t = 0; t < NT; ++t)
      F.b[t] =
          load_frag16(Brow[t] + k0 + h * 8, Brow[t] + k0 + 16 + h * 8);
  };
  auto STEP = [&](const FragSet<MT, NT>& F) {
#pragma unroll
    for (int mi = 0; mi < MT; ++mi)
#pragma unroll
      for (int t = 0; t < NT; ++t)
        acc[mi][t] = __builtin_amdgcn_wmma_f32_16x16x32_f16(
            false, F.a[mi], false, F.b[t], (short)0, acc[mi][t], false,
            false);
  };

  constexpr int STEPS = KDIM / 32;  // even for both instantiations
  static_assert(STEPS % 2 == 0, "K must be a multiple of 64");

  FragSet<MT, NT> P, Q;
  LOAD(P, 0);
  int s = 1;
  for (; s + 1 < STEPS; s += 2) {
    LOAD(Q, s * 32);        // in flight during STEP(P)
    STEP(P);
    LOAD(P, (s + 1) * 32);  // in flight during STEP(Q)
    STEP(Q);
  }
  LOAD(Q, (STEPS - 1) * 32);  // s == STEPS-1 here
  STEP(P);
  STEP(Q);

  // Writeback: C layout -> lane holds column n = r, rows h*8 + i
#pragma unroll
  for (int mi = 0; mi < MT; ++mi) {
#pragma unroll
    for (int t = 0; t < NT; ++t) {
      int n = (g * NT + t) * 16 + r;
      float bv = bias[n];
#pragma unroll
      for (int i = 0; i < 8; ++i) {
        float v = acc[mi][t][i] + bv;
        if (RELU) v = v > 0.f ? v : 0.f;
        C[(size_t)((mg * MT + mi) * 16 + h * 8 + i) * Ncols + n] = v;
      }
    }
  }
}

// ---------------------------------------------------------------------------
// RoI align (7x7 bilinear) + mean + NOTRACK zeroing.
// grid = B*N blocks; 256 threads = one channel each. feat stays hot in L2.
// ---------------------------------------------------------------------------
__global__ __launch_bounds__(256) void k_roi(const float* __restrict__ feat,
                                             const float* __restrict__ boxes,
                                             _Float16* __restrict__ obj) {
  const int m = blockIdx.x;    // b*16 + n
  const int c = threadIdx.x;   // channel
  const float* box = boxes + m * 4;
  float ymin = box[0], xmin = box[1], ymax = box[2], xmax = box[3];
  bool nonempty =
      (ymin != -1.f) || (xmin != -1.f) || (ymax != -1.f) || (xmax != -1.f);

  float accv = 0.f;
  if (nonempty) {
    const float* fb = feat + (size_t)(m >> 4) * FH * FW * CC;
#pragma unroll
    for (int sy = 0; sy < SROI; ++sy) {
      float py =
          (ymin + (ymax - ymin) * ((sy + 0.5f) * (1.f / SROI))) * FH - 0.5f;
      float y0f = floorf(py);
      float wy = py - y0f;
      int yi = (int)y0f;
      int y0 = min(max(yi, 0), FH - 1);
      int y1 = min(max(yi + 1, 0), FH - 1);
#pragma unroll
      for (int sx = 0; sx < SROI; ++sx) {
        float px =
            (xmin + (xmax - xmin) * ((sx + 0.5f) * (1.f / SROI))) * FW - 0.5f;
        float x0f = floorf(px);
        float wx = px - x0f;
        int xi = (int)x0f;
        int x0 = min(max(xi, 0), FW - 1);
        int x1 = min(max(xi + 1, 0), FW - 1);
        float f00 = fb[((y0 * FW) + x0) * CC + c];
        float f01 = fb[((y0 * FW) + x1) * CC + c];
        float f10 = fb[((y1 * FW) + x0) * CC + c];
        float f11 = fb[((y1 * FW) + x1) * CC + c];
        float top = f00 + wx * (f01 - f00);
        float bot = f10 + wx * (f11 - f10);
        accv += top + wy * (bot - top);
      }
    }
    accv *= (1.f / 49.f);
  }
  obj[(size_t)m * CC + c] = (_Float16)accv;
}

// ---------------------------------------------------------------------------
// Workspace layout (bytes):
//   A    f16 [16384,768]  @ 0          (25,165,824)
//   BT   f16 [256,768]    @ 25165824   (   393,216)
//   feat f32 [16384,256]  @ 25559040   (16,777,216)
//   obj  f16 [256,256]    @ 42336256   (   131,072)
//   DT   f16 [256,256]    @ 42467328   (   131,072)   total ~42.6 MB
// ---------------------------------------------------------------------------
extern "C" void kernel_launch(void* const* d_in, const int* in_sizes, int n_in,
                              void* d_out, int out_size, void* d_ws,
                              size_t ws_size, hipStream_t stream) {
  (void)in_sizes; (void)n_in; (void)out_size; (void)ws_size;
  const float* images  = (const float*)d_in[0];
  const float* bboxes  = (const float*)d_in[1];
  const float* conv_w  = (const float*)d_in[2];
  const float* conv_b  = (const float*)d_in[3];
  const float* dense_w = (const float*)d_in[4];
  const float* dense_b = (const float*)d_in[5];

  char* ws = (char*)d_ws;
  _Float16* A    = (_Float16*)(ws + 0);
  _Float16* BT   = (_Float16*)(ws + 25165824);
  float*    feat = (float*)   (ws + 25559040);
  _Float16* obj  = (_Float16*)(ws + 42336256);
  _Float16* DT   = (_Float16*)(ws + 42467328);
  float* out = (float*)d_out;

  // Stage 0: layout prep (f32 -> f16, weight transposes)
  k_im2col<<<(MCONV * KCONV) / 256, 256, 0, stream>>>(images, A);
  k_conv_w_t<<<(CC * KCONV + 255) / 256, 256, 0, stream>>>(conv_w, BT);
  k_dense_w_t<<<(DD * CC + 255) / 256, 256, 0, stream>>>(dense_w, DT);

  // Stage 1: conv stem GEMM (M=16384, N=256, K=768), bias + ReLU
  {
    const int MGroups = (MCONV / 16) / 2;   // 512 (2 M-tiles per wave)
    const int NGroups = (CC / 16) / 4;      // 4  (4 N-tiles per wave)
    const int waves = MGroups * NGroups;    // 2048 waves
    k_gemm<KCONV, 2, 4, true><<<(waves + 7) / 8, 256, 0, stream>>>(
        A, BT, conv_b, feat, MGroups, NGroups, CC);
  }

  // Stage 2: RoI align + mean + NOTRACK masking -> obj f16
  k_roi<<<BB * NBOX, CC, 0, stream>>>(feat, bboxes, obj);

  // Stage 3: dense head GEMM (M=256, N=256, K=256), bias, no ReLU -> d_out
  {
    const int MGroups = ((BB * NBOX) / 16) / 2;  // 8
    const int NGroups = (DD / 16) / 4;           // 4
    const int waves = MGroups * NGroups;         // 32 waves
    k_gemm<CC, 2, 4, false><<<(waves + 7) / 8, 256, 0, stream>>>(
        obj, DT, dense_b, out, MGroups, NGroups, DD);
  }
}
